// _NonLocalBlockND_42880953483837
// MI455X (gfx1250) — matrix operand order repaired and checked by reading, not compiled, not run on hardware
//
#include <hip/hip_runtime.h>
#include <hip/hip_bf16.h>

// ---- problem constants ----
#define Bn   4
#define Cc   256
#define Dd   128              // INTER
#define Hh   96
#define Ww   96
#define Nn   (Hh*Ww)          // 9216
#define Mm   ((Hh/2)*(Ww/2))  // 2304

typedef __attribute__((ext_vector_type(16))) __bf16 v16bf;
typedef __attribute__((ext_vector_type(8)))  float  v8f;

// WMMA helper: D = A(16x32 bf16) * B(32x16 bf16) + C(16x16 f32)
__device__ __forceinline__ v8f wmma_bf(v16bf a, v16bf b, v8f c) {
  return __builtin_amdgcn_wmma_f32_16x16x32_bf16(false, a, false, b,
                                                 (short)0, c, false, false);
}

// A-operand load: lane holds row m=lane%16; 16 contiguous elems at p[0..15]
// (K=16h+i). p must be 16B aligned.
__device__ __forceinline__ v16bf load_a16(const __bf16* p) {
  union { uint4 u[2]; v16bf v; } t;
  t.u[0] = *(const uint4*)(p);
  t.u[1] = *(const uint4*)(p + 8);
  return t.v;
}

// ---- async global->LDS copy (CDNA5 GLOBAL_LOAD_ASYNC_TO_LDS_B128), guarded ----
#if defined(__AMDGCN__) && __has_builtin(__builtin_amdgcn_global_load_async_to_lds_b128)
#define HAVE_ASYNC_LDS 1
#endif

typedef int vi4 __attribute__((__vector_size__(16)));
typedef __attribute__((address_space(1))) vi4* gptr4;   // global int4*
typedef __attribute__((address_space(3))) vi4* lptr4;   // LDS int4*

__device__ __forceinline__ void async_cp16(__bf16* lds, const __bf16* glb) {
#ifdef HAVE_ASYNC_LDS
  __builtin_amdgcn_global_load_async_to_lds_b128(
      (gptr4)(uintptr_t)glb,
      (lptr4)(unsigned int)(uintptr_t)lds,
      0, 0);
#else
  *(uint4*)lds = *(const uint4*)glb;
#endif
}

__device__ __forceinline__ void wait_async_lds() {
#ifdef HAVE_ASYNC_LDS
#if __has_builtin(__builtin_amdgcn_s_wait_asynccnt)
  __builtin_amdgcn_s_wait_asynccnt(0);
#else
  asm volatile("s_wait_asynccnt 0" ::: "memory");
#endif
#endif
}

// ---------------------------------------------------------------------------
// Kernel 0: convert weights to bf16 (packed A-operand matrices)
// wbf: (3,128,256) = {theta_w, phi_w, g_w};  wwbf: (256,128) = W_w
__global__ __launch_bounds__(256) void k_wcvt(
    const float* __restrict__ tw, const float* __restrict__ pw,
    const float* __restrict__ gw, const float* __restrict__ Wwm,
    __bf16* __restrict__ wbf, __bf16* __restrict__ wwbf) {
  int i = blockIdx.x * 256 + threadIdx.x;
  if (i < Dd * Cc) {
    wbf[i]           = (__bf16)tw[i];
    wbf[Dd*Cc + i]   = (__bf16)pw[i];
    wbf[2*Dd*Cc + i] = (__bf16)gw[i];
  }
  if (i < Cc * Dd) wwbf[i] = (__bf16)Wwm[i];
}

// ---------------------------------------------------------------------------
// Kernel 1: fused theta/phi/g 1x1 convs.  Grid: Bn*(Nn/32) blocks x 256 thr.
// 32-pixel tiles: each weight A-operand feeds TWO WMMAs (1 load per wmma).
// theta -> bf16 (B,N,128);  phi,g -> f32 (B,128,N)
__global__ __launch_bounds__(256) void k_conv3(
    const float* __restrict__ x,        // (B,C,N)
    const __bf16* __restrict__ wbf,     // (3,128,256)
    const float* __restrict__ tb, const float* __restrict__ pb,
    const float* __restrict__ gb,
    __bf16* __restrict__ theta_bf,      // (B,N,128)
    float* __restrict__ phi_f,          // (B,128,N)
    float* __restrict__ g_f) {          // (B,128,N)
  __shared__ __attribute__((aligned(32))) __bf16 sX[32][Cc]; // [col][c] 16KB
  const int blk = blockIdx.x;
  const int b   = blk / (Nn / 32);
  const int n0  = (blk % (Nn / 32)) * 32;
  const int tid = threadIdx.x;
  // cooperative stage of x tile (256 channels x 32 pixels), f32 -> bf16
  const int col  = tid & 31;
  const int crow = tid >> 5;
  for (int k = 0; k < 32; ++k) {
    int c = crow + 8 * k;
    sX[col][c] = (__bf16)x[((size_t)b * Cc + c) * Nn + n0 + col];
  }
  __syncthreads();
  const int wave = tid >> 5, lane = tid & 31;
  const int half = lane >> 4, l16 = lane & 15;
  const int o0 = wave * 16;

  for (int t = 0; t < 3; ++t) {      // 0=theta 1=phi 2=g (o-tile = wave)
    const __bf16* wbase = wbf + ((size_t)t * Dd + o0 + l16) * Cc + 16 * half;
    v16bf aCur = load_a16(wbase);
    v8f acc0 = {}, acc1 = {};
    for (int c = 0; c < 8; ++c) {    // software-pipelined A loads
      v16bf aNxt = (c < 7) ? load_a16(wbase + 32 * (c + 1)) : aCur;
      acc0 = wmma_bf(aCur, *(const v16bf*)(&sX[l16][32 * c + 16 * half]), acc0);
      acc1 = wmma_bf(aCur, *(const v16bf*)(&sX[16 + l16][32 * c + 16 * half]), acc1);
      aCur = aNxt;
    }
    const float* bias = (t == 0) ? tb : (t == 1) ? pb : gb;
    if (t == 0) {
      union { uint4 u; __bf16 h[8]; } p0, p1;
      for (int j = 0; j < 8; ++j) {
        float bv = bias[o0 + 8 * half + j];
        p0.h[j] = (__bf16)(acc0[j] + bv);
        p1.h[j] = (__bf16)(acc1[j] + bv);
      }
      *(uint4*)(theta_bf + ((size_t)b * Nn + n0 + l16) * Dd + o0 + 8 * half) = p0.u;
      *(uint4*)(theta_bf + ((size_t)b * Nn + n0 + 16 + l16) * Dd + o0 + 8 * half) = p1.u;
    } else {
      float* dst = (t == 1) ? phi_f : g_f;
      for (int j = 0; j < 8; ++j) {
        int o = o0 + 8 * half + j;
        float bv = bias[o];
        size_t row = ((size_t)b * Dd + o) * Nn + n0;
        dst[row + l16]      = acc0[j] + bv;
        dst[row + 16 + l16] = acc1[j] + bv;
      }
    }
  }
}

// ---------------------------------------------------------------------------
// Kernel 2: 2x2 maxpool; phi -> bf16 (B,M,128), g -> bf16 (B,128,M)
__global__ __launch_bounds__(256) void k_pool(
    const float* __restrict__ phi_f, const float* __restrict__ g_f,
    __bf16* __restrict__ phi_p, __bf16* __restrict__ g_p) {
  int idx = blockIdx.x * 256 + threadIdx.x;
  if (idx >= Bn * Dd * Mm) return;
  const int m  = idx % Mm;
  const int d  = (idx / Mm) % Dd;
  const int b  = idx / (Mm * Dd);
  const int mh = m / (Ww / 2), mw = m % (Ww / 2);
  const size_t base = ((size_t)b * Dd + d) * Nn + (size_t)(2 * mh) * Ww + 2 * mw;
  float pv = fmaxf(fmaxf(phi_f[base], phi_f[base + 1]),
                   fmaxf(phi_f[base + Ww], phi_f[base + Ww + 1]));
  float gv = fmaxf(fmaxf(g_f[base], g_f[base + 1]),
                   fmaxf(g_f[base + Ww], g_f[base + Ww + 1]));
  phi_p[((size_t)b * Mm + m) * Dd + d] = (__bf16)pv;
  g_p[((size_t)b * Dd + d) * Mm + m]   = (__bf16)gv;
}

// ---------------------------------------------------------------------------
// Kernel 3: flash-style attention, Mc=64 chunks, 128 thr (4 waves).
// Double-buffered async copy of phi chunks into LDS; all waves compute S;
// softmax parallelized across all 128 lanes.
__global__ __launch_bounds__(128) void k_attn(
    const __bf16* __restrict__ theta_bf,  // (B,N,128)
    const __bf16* __restrict__ phi_p,     // (B,M,128)
    const __bf16* __restrict__ g_p,       // (B,128,M)
    __bf16* __restrict__ y_bf) {          // (B,N,128)
  __shared__ __attribute__((aligned(32))) __bf16 sPhi[2][64][Dd];  // 2x16KB
  __shared__ __attribute__((aligned(32))) float  sS[16][64];       // 4KB
  __shared__ __attribute__((aligned(32))) __bf16 sP[16][64];       // 2KB
  __shared__ float sPart[16][8];
  __shared__ float sMax[16], sSum[16], sScale[16];

  const int blk = blockIdx.x;
  const int b   = blk / (Nn / 16);
  const int n0  = (blk % (Nn / 16)) * 16;
  const int tid = threadIdx.x;
  const int wave = tid >> 5, lane = tid & 31;
  const int half = lane >> 4, l16 = lane & 15;
  const int rS = tid & 15, qS = tid >> 4;   // softmax: row, 8-col group

  const __bf16* phiBase = phi_p + (size_t)b * Mm * Dd;
  auto stage = [&](int buf, int mc) {       // 16KB contiguous chunk -> LDS
    const __bf16* src = phiBase + (size_t)mc * Dd;
    __bf16* dst = &sPhi[buf][0][0];
    for (int i = 0; i < 8; ++i) {
      int e = (tid + 128 * i) * 8;          // 16B per thread per iter
      async_cp16(dst + e, src + e);
    }
  };

  // preload theta A tiles (16 rows x 128 K) - used by every wave
  v16bf aTh[4];
  const __bf16* thb = theta_bf + ((size_t)b * Nn + n0 + l16) * Dd;
  for (int c = 0; c < 4; ++c) aTh[c] = load_a16(thb + 32 * c + 16 * half);

  stage(0, 0);
  if (tid < 16) { sMax[tid] = -3.0e38f; sSum[tid] = 0.f; }
  v8f oA = {}, oB = {};   // this wave owns d-cols [32*wave, 32*wave+32)
  wait_async_lds();
  __syncthreads();

  for (int ci = 0; ci < Mm / 64; ++ci) {
    const int mc  = ci * 64;
    const int buf = ci & 1;
    if (ci + 1 < Mm / 64) stage(buf ^ 1, mc + 64);   // async prefetch next

    // ---- S tile: wave w computes cols [16w,16w+16) of the 64-col chunk ----
    {
      v8f s = {};
      const __bf16* pb = &sPhi[buf][16 * wave + l16][0];
      for (int c = 0; c < 4; ++c)
        s = wmma_bf(aTh[c], *(const v16bf*)(pb + 32 * c + 16 * half), s);
      for (int j = 0; j < 8; ++j) sS[8 * half + j][16 * wave + l16] = s[j];
    }
    __syncthreads();

    // ---- online softmax, parallel over 128 lanes (8 elems each) ----
    {
      float mx = sS[rS][8 * qS];
      for (int j = 1; j < 8; ++j) mx = fmaxf(mx, sS[rS][8 * qS + j]);
      sPart[rS][qS] = mx;
    }
    __syncthreads();
    if (tid < 16) {
      float old = sMax[tid], mx = old;
      for (int q = 0; q < 8; ++q) mx = fmaxf(mx, sPart[tid][q]);
      sMax[tid]   = mx;
      sScale[tid] = __expf(old - mx);
    }
    __syncthreads();
    {
      float mx = sMax[rS], sum = 0.f;
      for (int j = 0; j < 8; ++j) {
        float p = __expf(sS[rS][8 * qS + j] - mx);
        sP[rS][8 * qS + j] = (__bf16)p;
        sum += p;
      }
      sPart[rS][qS] = sum;
    }
    __syncthreads();
    if (tid < 16) {   // running denominator (disjoint LDS vs O-update below)
      float s2 = 0.f;
      for (int q = 0; q < 8; ++q) s2 += sPart[tid][q];
      sSum[tid] = sSum[tid] * sScale[tid] + s2;
    }

    // ---- O update: rescale accumulators, P(16x64) @ g(64x32) ----
    for (int j = 0; j < 8; ++j) {
      float sc = sScale[8 * half + j];
      oA[j] *= sc; oB[j] *= sc;
    }
    v16bf pA0 = load_a16(&sP[l16][16 * half]);        // K = m-local 0..31
    v16bf pA1 = load_a16(&sP[l16][32 + 16 * half]);   // K = m-local 32..63
    const int d0 = 32 * wave;
    const __bf16* gr0 = g_p + ((size_t)b * Dd + d0 + l16) * Mm + mc + 16 * half;
    const __bf16* gr1 = g_p + ((size_t)b * Dd + d0 + 16 + l16) * Mm + mc + 16 * half;
    oA = wmma_bf(pA0, *(const v16bf*)(gr0), oA);
    oA = wmma_bf(pA1, *(const v16bf*)(gr0 + 32), oA);
    oB = wmma_bf(pA0, *(const v16bf*)(gr1), oB);
    oB = wmma_bf(pA1, *(const v16bf*)(gr1 + 32), oB);

    wait_async_lds();   // prefetch for next chunk complete before barrier
    __syncthreads();
  }

  for (int j = 0; j < 8; ++j) {
    float inv = 1.0f / sSum[8 * half + j];
    __bf16* dst = y_bf + ((size_t)b * Nn + n0 + 8 * half + j) * Dd + 32 * wave;
    dst[l16]      = (__bf16)(oA[j] * inv);
    dst[16 + l16] = (__bf16)(oB[j] * inv);
  }
}

// ---------------------------------------------------------------------------
// Kernel 4: W 1x1 conv + BatchNorm + residual, 32-pixel tiles.
// Grid: Bn*(Nn/32) x 256 thr.  f32 out (B,C,H,W)
__global__ __launch_bounds__(256) void k_wconv(
    const __bf16* __restrict__ y_bf,   // (B,N,128)
    const __bf16* __restrict__ wwbf,   // (256,128)
    const float* __restrict__ Wb,
    const float* __restrict__ gma, const float* __restrict__ bta,
    const float* __restrict__ mean, const float* __restrict__ var,
    const float* __restrict__ x, float* __restrict__ out) {
  const int blk = blockIdx.x;
  const int b   = blk / (Nn / 32);
  const int n0  = (blk % (Nn / 32)) * 32;
  const int tid = threadIdx.x;
  const int wave = tid >> 5, lane = tid & 31;
  const int half = lane >> 4, l16 = lane & 15;
  // y tile B-operands for both 16-col sets (shared by both c-tiles)
  v16bf yB0[4], yB1[4];
  const __bf16* yb0 = y_bf + ((size_t)b * Nn + n0 + l16) * Dd;
  const __bf16* yb1 = y_bf + ((size_t)b * Nn + n0 + 16 + l16) * Dd;
  for (int c = 0; c < 4; ++c) {
    yB0[c] = *(const v16bf*)(yb0 + 32 * c + 16 * half);
    yB1[c] = *(const v16bf*)(yb1 + 32 * c + 16 * half);
  }
  for (int ct = wave; ct < 16; ct += 8) {
    const int c0 = ct * 16;
    const __bf16* wb = wwbf + (size_t)(c0 + l16) * Dd + 16 * half;
    v16bf aCur = load_a16(wb);
    v8f acc0 = {}, acc1 = {};
    for (int c = 0; c < 4; ++c) {    // software-pipelined A loads
      v16bf aNxt = (c < 3) ? load_a16(wb + 32 * (c + 1)) : aCur;
      acc0 = wmma_bf(aCur, yB0[c], acc0);
      acc1 = wmma_bf(aCur, yB1[c], acc1);
      aCur = aNxt;
    }
    for (int j = 0; j < 8; ++j) {
      int ch = c0 + 8 * half + j;
      float inv  = gma[ch] * rsqrtf(var[ch] + 1e-5f);
      float bias = (Wb[ch] - mean[ch]) * inv + bta[ch];
      size_t off = ((size_t)b * Cc + ch) * Nn + n0 + l16;
      out[off]      = acc0[j] * inv + bias + x[off];
      out[off + 16] = acc1[j] * inv + bias + x[off + 16];
    }
  }
}

// ---------------------------------------------------------------------------
extern "C" void kernel_launch(void* const* d_in, const int* in_sizes, int n_in,
                              void* d_out, int out_size, void* d_ws, size_t ws_size,
                              hipStream_t stream) {
  const float* x       = (const float*)d_in[0];
  const float* theta_w = (const float*)d_in[1];
  const float* theta_b = (const float*)d_in[2];
  const float* phi_w   = (const float*)d_in[3];
  const float* phi_b   = (const float*)d_in[4];
  const float* g_w     = (const float*)d_in[5];
  const float* g_b     = (const float*)d_in[6];
  const float* W_w     = (const float*)d_in[7];
  const float* W_b     = (const float*)d_in[8];
  const float* bn_g    = (const float*)d_in[9];
  const float* bn_b    = (const float*)d_in[10];
  const float* bn_m    = (const float*)d_in[11];
  const float* bn_v    = (const float*)d_in[12];
  float* out = (float*)d_out;

  // workspace carve-up (256B aligned sub-buffers)
  size_t off = 0;
  auto carve = [&](size_t bytes) -> void* {
    void* p = (char*)d_ws + off;
    off += (bytes + 255) & ~(size_t)255;
    return p;
  };
  __bf16* wbf      = (__bf16*)carve((size_t)3 * Dd * Cc * 2);
  __bf16* wwbf     = (__bf16*)carve((size_t)Cc * Dd * 2);
  __bf16* theta_bf = (__bf16*)carve((size_t)Bn * Nn * Dd * 2);
  float*  phi_f    = (float*) carve((size_t)Bn * Dd * Nn * 4);
  float*  g_f      = (float*) carve((size_t)Bn * Dd * Nn * 4);
  __bf16* phi_p    = (__bf16*)carve((size_t)Bn * Mm * Dd * 2);
  __bf16* g_p      = (__bf16*)carve((size_t)Bn * Dd * Mm * 2);
  __bf16* y_bf     = (__bf16*)carve((size_t)Bn * Nn * Dd * 2);
  if (off > ws_size) return;  // workspace too small -> no-op (deterministic)

  const int tiles16 = Bn * (Nn / 16);                     // 2304
  const int tiles32 = Bn * (Nn / 32);                     // 1152
  k_wcvt <<<(Dd * Cc + 255) / 256, 256, 0, stream>>>(theta_w, phi_w, g_w, W_w,
                                                     wbf, wwbf);
  k_conv3<<<tiles32, 256, 0, stream>>>(x, wbf, theta_b, phi_b, g_b,
                                       theta_bf, phi_f, g_f);
  k_pool <<<(Bn * Dd * Mm + 255) / 256, 256, 0, stream>>>(phi_f, g_f,
                                                          phi_p, g_p);
  k_attn <<<tiles16, 128, 0, stream>>>(theta_bf, phi_p, g_p, y_bf);
  k_wconv<<<tiles32, 256, 0, stream>>>(y_bf, wwbf, W_b, bn_g, bn_b, bn_m, bn_v,
                                       x, out);
}
